// GraphAttention_78975858639656
// MI455X (gfx1250) — matrix operand order repaired and checked by reading, not compiled, not run on hardware
//
#include <hip/hip_runtime.h>

#define B_    4
#define N_    2048
#define F_    64
#define FP_   32
#define H_    4
#define HF_   (H_ * FP_)
#define TILES_ (N_ / 16)
#define BN_EPS 1e-3f

typedef _Float16 half_t;
typedef __attribute__((ext_vector_type(16))) _Float16 v16h;
typedef __attribute__((ext_vector_type(8)))  float    v8f;

// ---------------------------------------------------------------------------
// Kernel 1: feats = einsum('bnf,hfd->bhnd', x, W)  (stored f16, d-contiguous)
//           s = feats . a_self,  t = feats . a_neigh  (f32)
// One thread per (b,h,n). ~0.14 GFLOP total -> trivial.
// ---------------------------------------------------------------------------
__global__ void gat_feats_kernel(const float* __restrict__ x,
                                 const float* __restrict__ W,
                                 const float* __restrict__ a_self,
                                 const float* __restrict__ a_neigh,
                                 half_t* __restrict__ feats_h,
                                 float* __restrict__ s,
                                 float* __restrict__ t) {
  int gid = blockIdx.x * blockDim.x + threadIdx.x;
  if (gid >= B_ * H_ * N_) return;
  int n  = gid % N_;
  int bh = gid / N_;        // b*H + h
  int h  = bh % H_;
  const float* xr = x + ((size_t)(bh / H_) * N_ + n) * F_;
  float sacc = 0.f, tacc = 0.f;
  for (int d = 0; d < FP_; ++d) {
    float acc = 0.f;
#pragma unroll
    for (int k = 0; k < F_; ++k)
      acc += xr[k] * W[((size_t)h * F_ + k) * FP_ + d];
    feats_h[((size_t)bh * N_ + n) * FP_ + d] = (half_t)acc;
    sacc += acc * a_self[h * FP_ + d];
    tacc += acc * a_neigh[h * FP_ + d];
  }
  s[(size_t)bh * N_ + n] = sacc;
  t[(size_t)bh * N_ + n] = tacc;
}

// ---------------------------------------------------------------------------
// Kernel 2: fused attention (softmax * adj) @ feats via WMMA.
// Two passes only: (1) row max, (2) UNNORMALIZED exp*adj accumulated into the
// WMMA C fragments while Z is tracked; the 1/Z row scale is applied in the
// epilogue (softmax normalization is a per-row scalar). One exp per element.
// One wave (32 lanes) owns a 16-row output tile of one (b,h).
// ---------------------------------------------------------------------------
__global__ void __launch_bounds__(256)
gat_attn_wmma_kernel(const float* __restrict__ adj,
                     const float* __restrict__ mask,
                     const half_t* __restrict__ feats_h,
                     const float* __restrict__ s,
                     const float* __restrict__ t,
                     const float* __restrict__ bias,
                     const float* __restrict__ gamma,
                     const float* __restrict__ beta,
                     const float* __restrict__ mean,
                     const float* __restrict__ var,
                     float* __restrict__ out,
                     float* __restrict__ loss) {
  const int tid  = blockIdx.x * blockDim.x + threadIdx.x;
  const int wave = tid >> 5;
  const int lane = tid & 31;

  const int tile = wave % TILES_;
  const int bh   = wave / TILES_;      // b*H + h
  const int h    = bh % H_;
  const int b    = bh / H_;
  const int i0   = tile * 16;

  const int mrow = lane & 15;          // output row within tile (A-matrix M)
  const int irow = i0 + mrow;
  const int kb   = (lane >> 4) << 3;   // K base: 0 for lanes 0-15, 8 for 16-31

  const float* trow    = t    + (size_t)bh * N_;
  const float* adjrow  = adj  + ((size_t)b * N_ + irow) * N_;
  const float* maskrow = mask + ((size_t)b * N_ + irow) * N_;
  const float  sv      = s[(size_t)bh * N_ + irow];

  // ---- pass 1: row max of dense = leaky(s_i + t_j) + mask ----
  float m = -3.0e38f;
  for (int j0 = 0; j0 < N_; j0 += 32) {
#pragma unroll
    for (int e = 0; e < 16; ++e) {
      int k = kb + e + (e >= 8 ? 8 : 0);           // CDNA5 f16 A-matrix K map
      int j = j0 + k;
      float d = sv + trow[j];
      d = d > 0.f ? d : 0.2f * d;
      d += maskrow[j];
      m = fmaxf(m, d);
    }
  }
  m = fmaxf(m, __shfl_xor(m, 16, 32));             // combine lane pair (wave32)

  // ---- pass 2: unnormalized alpha tile x feats via WMMA, Z + stats fused ----
  v8f c0 = {};                                     // cols d = 0..15
  v8f c1 = {};                                     // cols d = 16..31
  float Z = 0.f, cnt = 0.f, sa = 0.f, dg = 0.f;

#pragma unroll 2
  for (int j0 = 0; j0 < N_; j0 += 32) {
    __builtin_prefetch(adjrow + j0 + 32, 0, 1);    // global_prefetch_b8 (spec.)
    v16h a;
#pragma unroll
    for (int e = 0; e < 16; ++e) {
      int k = kb + e + (e >= 8 ? 8 : 0);
      int j = j0 + k;
      float d = sv + trow[j];
      d = d > 0.f ? d : 0.2f * d;
      d += maskrow[j];
      float p    = __expf(d - m);                  // <= 1, f16-safe
      float adjv = adjrow[j];
      float al   = p * adjv;                       // unnormalized alpha
      Z   += p;                                    // softmax denominator
      cnt += (al   != 0.f) ? 1.f : 0.f;            // nonz (scale-invariant)
      sa  += al;                                   // unnormalized sum |alpha|
      dg  += (adjv != 0.f) ? 1.f : 0.f;            // deg
      a[e] = (half_t)al;
    }
    // B fragment: lane = column d (lane&15), elements over K in A's K map
    v16h b0, b1;
#pragma unroll
    for (int e = 0; e < 16; ++e) {
      int k = kb + e + (e >= 8 ? 8 : 0);
      const half_t* frow = feats_h + ((size_t)bh * N_ + j0 + k) * FP_;
      b0[e] = frow[mrow];
      b1[e] = frow[16 + mrow];
    }
    c0 = __builtin_amdgcn_wmma_f32_16x16x32_f16(false, a, false, b0,
                                                (short)0, c0, false, false);
    c1 = __builtin_amdgcn_wmma_f32_16x16x32_f16(false, a, false, b1,
                                                (short)0, c1, false, false);
  }

  // ---- per-row combine across the lane pair; Z lives on lane = row ----
  Z   += __shfl_xor(Z,   16, 32);
  cnt += __shfl_xor(cnt, 16, 32);
  sa  += __shfl_xor(sa,  16, 32);
  dg  += __shfl_xor(dg,  16, 32);
  const float invZ = 1.0f / Z;                     // this lane's row

  // ---- attention-stats losses ----
  float u  = cnt - dg;
  float eL = sa * invZ;                            // normalized sum alpha
#pragma unroll
  for (int off = 8; off > 0; off >>= 1) {
    u  += __shfl_xor(u,  off, 32);
    eL += __shfl_xor(eL, off, 32);
  }
  if (lane == 0) {
    atomicAdd(loss + 0, u  * (1.0f / N_));
    atomicAdd(loss + 1, eL * (1.0f / N_));
  }

  // ---- epilogue: row-scale by 1/Z, + bias, inference BN, ReLU ----
  const int mbase = (lane >> 4) << 3;              // C-frag row base 0 / 8
  const int ci0 = h * FP_ + mrow;                  // concat channel, frag 0
  const int ci1 = h * FP_ + 16 + mrow;             // concat channel, frag 1
  const float sc0 = rsqrtf(var[ci0] + BN_EPS) * gamma[ci0];
  const float sc1 = rsqrtf(var[ci1] + BN_EPS) * gamma[ci1];
  const float b0v = bias[ci0], b1v = bias[ci1];
#pragma unroll
  for (int r = 0; r < 8; ++r) {
    int n = i0 + r + mbase;
    float invZr = __shfl(invZ, r + mbase, 32);     // row r+mbase's 1/Z
    float v0 = c0[r] * invZr + b0v;
    v0 = (v0 - mean[ci0]) * sc0 + beta[ci0];
    out[((size_t)b * N_ + n) * HF_ + ci0] = fmaxf(v0, 0.f);
    float v1 = c1[r] * invZr + b1v;
    v1 = (v1 - mean[ci1]) * sc1 + beta[ci1];
    out[((size_t)b * N_ + n) * HF_ + ci1] = fmaxf(v1, 0.f);
  }
}

// ---------------------------------------------------------------------------
__global__ void gat_loss_init(float* loss) {
  if (threadIdx.x < 2) loss[threadIdx.x] = 0.f;
}

extern "C" void kernel_launch(void* const* d_in, const int* in_sizes, int n_in,
                              void* d_out, int out_size, void* d_ws,
                              size_t ws_size, hipStream_t stream) {
  (void)in_sizes; (void)n_in; (void)out_size; (void)ws_size;
  const float* x       = (const float*)d_in[0];
  const float* adj     = (const float*)d_in[1];
  const float* mask    = (const float*)d_in[2];
  const float* W       = (const float*)d_in[3];
  const float* a_self  = (const float*)d_in[4];
  const float* a_neigh = (const float*)d_in[5];
  const float* bias    = (const float*)d_in[6];
  const float* gamma   = (const float*)d_in[7];
  const float* beta    = (const float*)d_in[8];
  const float* mmean   = (const float*)d_in[9];
  const float* mvar    = (const float*)d_in[10];

  float* out  = (float*)d_out;
  float* loss = out + (size_t)B_ * N_ * HF_;       // uloss, eloss

  char* ws = (char*)d_ws;
  half_t* feats_h = (half_t*)ws;                                   // 2 MB
  float*  s = (float*)(ws + (size_t)B_ * H_ * N_ * FP_ * sizeof(half_t));
  float*  t = s + (size_t)B_ * H_ * N_;

  gat_loss_init<<<1, 32, 0, stream>>>(loss);
  gat_feats_kernel<<<(B_ * H_ * N_ + 255) / 256, 256, 0, stream>>>(
      x, W, a_self, a_neigh, feats_h, s, t);
  // 2048 waves total = 65536 threads, 8 waves per block
  gat_attn_wmma_kernel<<<(B_ * H_ * TILES_ * 32) / 256, 256, 0, stream>>>(
      adj, mask, feats_h, s, t, bias, gamma, beta, mmean, mvar, out, loss);
}
